// binary_conv_35923106464284
// MI455X (gfx1250) — matrix operand order repaired and checked by reading, not compiled, not run on hardware
//
#include <hip/hip_runtime.h>

typedef __attribute__((ext_vector_type(16))) _Float16 v16h;
typedef __attribute__((ext_vector_type(8)))  _Float16 v8h;
typedef __attribute__((ext_vector_type(4)))  _Float16 v4h;
typedef __attribute__((ext_vector_type(8)))  float    v8f;

#define CI 128
#define CO 256
#define IH 112
#define IW 112
#define TS 16            // 16x16 spatial tile per workgroup
#define XP 136           // padded per-pixel pitch in f16 (272 B) -> bank-conflict-free A loads

union V16 { v16h v; v8h h[2]; };

// ---------------------------------------------------------------------------
// Prep: binarize weights (sign -> {-1,0,+1}, exact in f16) and swizzle into
// WMMA B-fragment blobs, chunk-major for conflict-minimal ds_load_b128:
//   blob b = (e*4 + kblk)*16 + nblk, 512 f16 each;
//   within blob: f16 offset = chunk*256 + lane*8 + e7   (chunk = elem>>3)
//   lane l holds column co = nblk*16 + (l&15); K = chunk*16 + (l>=16 ? 8 : 0) + e7.
// Flat thread index g == destination f16 offset.
// ---------------------------------------------------------------------------
__global__ __launch_bounds__(256)
void binconv_prep_weights(const float* __restrict__ w, _Float16* __restrict__ wb) {
    const int g     = blockIdx.x * 256 + threadIdx.x;   // 294912 total
    const int blob  = g >> 9;
    const int rem   = g & 511;
    const int chunk = rem >> 8;          // 0: elems 0..7, 1: elems 8..15
    const int lane  = (rem >> 3) & 31;
    const int e7    = rem & 7;
    const int elem  = chunk * 8 + e7;
    const int e     = blob >> 6;         // 0..8  (ky*3+kx)
    const int kblk  = (blob >> 4) & 3;   // ci block of 32
    const int nblk  = blob & 15;         // co block of 16
    const int ci    = kblk * 32 + chunk * 16 + ((lane & 16) ? 8 : 0) + e7;
    const int co    = nblk * 16 + (lane & 15);
    const float s   = w[((size_t)e * CI + ci) * CO + co];
    const _Float16 r = (s > 0.f) ? (_Float16)1.f
                                 : ((s < 0.f) ? (_Float16)-1.f : (_Float16)0.f);
    wb[g] = r;
}

// ---------------------------------------------------------------------------
// Main: implicit-GEMM binary conv via V_WMMA_F32_16X16X32_F16.
// grid = (7, 7, 32*8): tileX, tileY, z = n*8 + coblk (32 couts per WG).
// block = 256 threads = 8 waves; wave w owns spatial rows {2w, 2w+1}.
// Weights staged global->LDS with async b128 copies (ASYNCcnt).
// ---------------------------------------------------------------------------
__global__ __launch_bounds__(256)
void binconv_wmma(const float* __restrict__ x, const _Float16* __restrict__ wb,
                  float* __restrict__ out) {
    __shared__ __align__(16) _Float16 xs[(TS + 2) * (TS + 2) * XP]; // 88128 B
    __shared__ __align__(16) _Float16 lw[36 * 1024];                // 73728 B

    const int tid   = threadIdx.x;
    const int tileX = blockIdx.x;
    const int tileY = blockIdx.y;
    const int n     = blockIdx.z >> 3;
    const int coblk = blockIdx.z & 7;

    // ---- Async-stage this WG's weight slice (72 KB) into LDS -------------
    // src: pair p = (e*4+kb) at wb + (p*16 + coblk*2)*1024 bytes, 2048 B each.
    // dst: lw, identical ordering, 16 B chunks, 4608 chunks. (byte copy ->
    // layout-agnostic; overlaps with the VALU f32->f16 staging below.)
    {
        const unsigned ldsBase =
            (unsigned)(size_t)(&lw[0]);        // flat LDS addr truncates to LDS offset
        const unsigned long long gBase =
            (unsigned long long)(size_t)wb + (unsigned long long)(coblk * 2048);
        for (int c = tid; c < 4608; c += 256) {
            unsigned dst = ldsBase + (unsigned)c * 16u;
            unsigned long long src =
                gBase + (unsigned long long)((c >> 7) * 16384 + (c & 127) * 16);
            asm volatile("global_load_async_to_lds_b128 %0, %1, off"
                         :: "v"(dst), "v"(src) : "memory");
        }
    }

    // ---- Stage input tile (+SAME halo) fp32 -> f16 into LDS, float4 wide --
    const int nvec = (TS + 2) * (TS + 2) * CI / 4;  // 10368
    for (int i4 = tid; i4 < nvec; i4 += 256) {
        const int ci4 = (i4 & 31) * 4;
        const int pix = i4 >> 5;
        const int col = pix % (TS + 2);
        const int row = pix / (TS + 2);
        const int gy  = tileY * TS + row - 1;
        const int gx  = tileX * TS + col - 1;
        v4h h4 = {};
        if ((unsigned)gy < IH && (unsigned)gx < IW) {
            const float4 f = *(const float4*)&x[(((size_t)n * IH + gy) * IW + gx) * CI + ci4];
            h4[0] = (_Float16)f.x; h4[1] = (_Float16)f.y;
            h4[2] = (_Float16)f.z; h4[3] = (_Float16)f.w;
        }
        *(v4h*)&xs[(size_t)pix * XP + ci4] = h4;
    }

    asm volatile("s_wait_asynccnt 0x0" ::: "memory");
    __syncthreads();

    const int wave = tid >> 5;          // owns spatial rows 2*wave, 2*wave+1
    const int lane = tid & 31;
    const int m    = lane & 15;         // M: pixel x-offset within row
    const int hi   = lane >> 4;

    v8f acc00 = {}, acc01 = {};         // row0 x {co 0..15, co 16..31}
    v8f acc10 = {}, acc11 = {};         // row1 x {co 0..15, co 16..31}

    const _Float16* lwlane = lw + lane * 8;   // chunk-major blobs, 16 B lane stride

    #pragma unroll
    for (int e = 0; e < 9; ++e) {
        const int ky = e / 3, kx = e % 3;
        const int ab0 = ((wave * 2 + ky) * (TS + 2) + (m + kx)) * XP + hi * 8;
        const int ab1 = ab0 + (TS + 2) * XP;     // next spatial row
        #pragma unroll
        for (int kb = 0; kb < 4; ++kb) {
            V16 a0, a1, b0, b1;
            // A fragments: 16 px x 32 k per row (ds_load_b128 x2 each;
            // 272 B pixel pitch -> 2 requests/bank, the b128 minimum)
            const _Float16* ap0 = &xs[ab0 + kb * 32];
            const _Float16* ap1 = &xs[ab1 + kb * 32];
            a0.h[0] = *(const v8h*)(ap0);
            a0.h[1] = *(const v8h*)(ap0 + 16);
            a1.h[0] = *(const v8h*)(ap1);
            a1.h[1] = *(const v8h*)(ap1 + 16);
            // B fragments from LDS blobs (chunk-major: 16 B lane stride)
            const _Float16* bp = lwlane + (e * 4 + kb) * 1024;
            b0.h[0] = *(const v8h*)(bp);
            b0.h[1] = *(const v8h*)(bp + 256);
            b1.h[0] = *(const v8h*)(bp + 512);
            b1.h[1] = *(const v8h*)(bp + 768);
            acc00 = __builtin_amdgcn_wmma_f32_16x16x32_f16(
                        false, a0.v, false, b0.v, (short)0, acc00, false, false);
            acc01 = __builtin_amdgcn_wmma_f32_16x16x32_f16(
                        false, a0.v, false, b1.v, (short)0, acc01, false, false);
            acc10 = __builtin_amdgcn_wmma_f32_16x16x32_f16(
                        false, a1.v, false, b0.v, (short)0, acc10, false, false);
            acc11 = __builtin_amdgcn_wmma_f32_16x16x32_f16(
                        false, a1.v, false, b1.v, (short)0, acc11, false, false);
        }
    }

    // ---- Store D tiles: N (lane&15) = cout -> 64B-contiguous per half-wave.
    //      Non-temporal: out is write-once streaming; keep L2 for x/w reuse.
    const int gy0 = tileY * TS + wave * 2;
    const int gx0 = tileX * TS;
    const int co0 = coblk * 32 + (lane & 15);
    float* ob0 = out + (((size_t)n * IH + gy0) * IW + gx0) * CO + co0;
    float* ob1 = ob0 + (size_t)IW * CO;
    #pragma unroll
    for (int r = 0; r < 8; ++r) {
        const int px = r + hi * 8;
        __builtin_nontemporal_store(acc00[r], &ob0[(size_t)px * CO]);
        __builtin_nontemporal_store(acc01[r], &ob0[(size_t)px * CO + 16]);
        __builtin_nontemporal_store(acc10[r], &ob1[(size_t)px * CO]);
        __builtin_nontemporal_store(acc11[r], &ob1[(size_t)px * CO + 16]);
    }
}

// ---------------------------------------------------------------------------
extern "C" void kernel_launch(void* const* d_in, const int* in_sizes, int n_in,
                              void* d_out, int out_size, void* d_ws, size_t ws_size,
                              hipStream_t stream) {
    const float* x = (const float*)d_in[0];
    const float* w = (const float*)d_in[1];
    float* out     = (float*)d_out;
    _Float16* wb   = (_Float16*)d_ws;   // 576 KB of B-fragment blobs

    binconv_prep_weights<<<1152, 256, 0, stream>>>(w, wb);

    dim3 grid(IW / TS, IH / TS, 32 * 8);   // (7, 7, 256)
    binconv_wmma<<<grid, 256, 0, stream>>>(x, wb, out);
}
